// HybridBlock_15805479649520
// MI455X (gfx1250) — matrix operand order repaired and checked by reading, not compiled, not run on hardware
//
#include <hip/hip_runtime.h>
#include <math.h>

typedef __attribute__((ext_vector_type(16))) _Float16 v16h;
typedef __attribute__((ext_vector_type(8)))  float    v8f;
typedef __attribute__((ext_vector_type(4)))  float    f32x4;
typedef __attribute__((ext_vector_type(4)))  int      i32x4;

// ---- problem constants ----
constexpr int kB   = 16;
constexpr int kW   = 96;     // image width/height
constexpr int kN   = 9216;   // tokens = 96*96
constexpr int kDIM = 192;
constexpr int kC2  = 96;
constexpr int kNH  = 6;
constexpr int kHD  = 16;
constexpr int kIND = 12;     // FoPE in-dim
constexpr int kOUTD= 8;      // FoPE out-dim

// =====================================================================
// gfx1250 async global->LDS copies (ASYNCcnt path). Feature-gated so the
// file still compiles if the toolchain lacks the builtins.
// Probe-derived signatures: (as1 T* gsrc, as3 T* ldst, imm offset, imm cpol)
// with T = int (b32) / int4 (b128).
// =====================================================================
#if defined(__has_builtin)
#if __has_builtin(__builtin_amdgcn_global_load_async_to_lds_b32) && \
    __has_builtin(__builtin_amdgcn_s_wait_asynccnt)
#define HAVE_ASYNC_B32 1
#endif
#if __has_builtin(__builtin_amdgcn_global_load_async_to_lds_b128) && \
    __has_builtin(__builtin_amdgcn_s_wait_asynccnt)
#define HAVE_ASYNC_B128 1
#endif
#endif

__device__ __forceinline__ void cp_b32(const float* g, float* l) {
#if HAVE_ASYNC_B32
    __builtin_amdgcn_global_load_async_to_lds_b32(
        (__attribute__((address_space(1))) int*)g,
        (__attribute__((address_space(3))) int*)l, 0, 0);
#else
    *l = *g;
#endif
}
__device__ __forceinline__ void cp_b128(const float* g, float* l) {
#if HAVE_ASYNC_B128
    __builtin_amdgcn_global_load_async_to_lds_b128(
        (__attribute__((address_space(1))) i32x4*)g,
        (__attribute__((address_space(3))) i32x4*)l, 0, 0);
#else
    *(f32x4*)l = *(const f32x4*)g;
#endif
}
__device__ __forceinline__ void cp_wait() {
#if HAVE_ASYNC_B32 || HAVE_ASYNC_B128
    __builtin_amdgcn_s_wait_asynccnt(0);
#endif
}

// =====================================================================
// WMMA helpers (CDNA5 wave32 layouts, 16x16x32 f16 -> f32)
//   A (16x32, f16):  lane = half*16 + m ; vgpr j holds K = (j<4?0:16)+half*8+(j&3)*2 (+0/+1)
//   B (32x16, f16):  lane = half*16 + n ; vgpr j holds K = (j<4?0:16)+half*8+(j&3)*2 (+0/+1)
//   C/D (16x16,f32): lane: n = lane&15 ; vgpr r holds m = r + 8*(lane>>4)
// =====================================================================
__device__ __forceinline__ v16h ldA(const float* __restrict__ s, int ld) {
    int lane = threadIdx.x & 31;
    int m    = lane & 15;
    int half = lane >> 4;
    v16h a;
#pragma unroll
    for (int j = 0; j < 8; ++j) {
        int kb = (j < 4 ? 0 : 16) + half * 8 + (j & 3) * 2;
        a[2 * j]     = (_Float16)s[m * ld + kb];
        a[2 * j + 1] = (_Float16)s[m * ld + kb + 1];
    }
    return a;
}
// B stored k-major: element(k,n) = s[k*ld + n]
__device__ __forceinline__ v16h ldB(const float* __restrict__ s, int ld) {
    int lane = threadIdx.x & 31;
    int n    = lane & 15;
    int half = lane >> 4;
    v16h b;
#pragma unroll
    for (int j = 0; j < 8; ++j) {
        int kb = (j < 4 ? 0 : 16) + half * 8 + (j & 3) * 2;
        b[2 * j]     = (_Float16)s[kb * ld + n];
        b[2 * j + 1] = (_Float16)s[(kb + 1) * ld + n];
    }
    return b;
}
// B stored n-major (transposed weights): element(k,n) = s[n*ldn + k]
__device__ __forceinline__ v16h ldBt(const float* __restrict__ s, int ldn) {
    int lane = threadIdx.x & 31;
    int n    = lane & 15;
    int half = lane >> 4;
    v16h b;
#pragma unroll
    for (int j = 0; j < 8; ++j) {
        int kb = (j < 4 ? 0 : 16) + half * 8 + (j & 3) * 2;
        b[2 * j]     = (_Float16)s[n * ldn + kb];
        b[2 * j + 1] = (_Float16)s[n * ldn + kb + 1];
    }
    return b;
}
__device__ __forceinline__ void stD(float* __restrict__ d, int ld, v8f c) {
    int lane = threadIdx.x & 31;
    int n    = lane & 15;
    int half = lane >> 4;
#pragma unroll
    for (int r = 0; r < 8; ++r) d[(r + 8 * half) * ld + n] = c[r];
}
__device__ __forceinline__ v8f wmma16(v16h a, v16h b, v8f c) {
    return __builtin_amdgcn_wmma_f32_16x16x32_f16(false, a, false, b, (short)0, c,
                                                  false, false);
}

// =====================================================================
// hedgehog (16x16 matvec + softmax feature) + FoPE per token
// =====================================================================
__device__ __forceinline__ void hh_fope(const float* __restrict__ inp,     // 16 vals
                                        const float* __restrict__ w_s,     // LDS 16x16
                                        const float* __restrict__ b_s,     // LDS 16
                                        const float* __restrict__ fs_row,  // 16
                                        const float* __restrict__ fc_row,  // 16
                                        float* __restrict__ f,             // out 32 (pre-FoPE)
                                        float* __restrict__ ff) {          // out 32 (FoPE'd)
    float y[16];
#pragma unroll
    for (int j = 0; j < 16; ++j) {
        float acc = b_s[j];
#pragma unroll
        for (int i = 0; i < 16; ++i) acc += w_s[j * 16 + i] * inp[i];
        y[j] = acc;
    }
    float mx = 0.f;
#pragma unroll
    for (int j = 0; j < 16; ++j) mx = fmaxf(mx, fabsf(2.f * y[j]));
    float e2[16], em[16], ssum = 0.f;
#pragma unroll
    for (int j = 0; j < 16; ++j) {
        e2[j] = __expf(2.f * y[j] - mx);
        em[j] = __expf(-2.f * y[j] - mx);
        ssum += e2[j] + em[j];
    }
    float inv = 1.f / ssum;
#pragma unroll
    for (int j = 0; j < 16; ++j) {
        f[j]      = e2[j] * inv;
        f[16 + j] = em[j] * inv;
    }
#pragma unroll
    for (int j = 0; j < 32; ++j) {
        float fsv = fs_row[j & 15];
        float fcv = fc_row[j & 15];
        float rt  = (j < 16) ? -f[j + 16] : f[j - 16];
        ff[j]     = f[j] * fcv - rt * fsv;
    }
}

// =====================================================================
// K0: FoPE sin/cos mixing tables  fs_tab/fc_tab [NH][N][16]
// =====================================================================
__global__ void fope_tab_kernel(const float* __restrict__ sin_coef,
                                const float* __restrict__ cos_coef,
                                float* __restrict__ fs_tab,
                                float* __restrict__ fc_tab) {
    int idx = blockIdx.x * blockDim.x + threadIdx.x;
    if (idx >= kNH * kN) return;
    int h = idx / kN, t = idx % kN;
    float sv[kIND], cv[kIND];
#pragma unroll
    for (int i = 0; i < kIND; ++i) {
        // inv_freq[i] = 10000^(-i/16)
        float fr = __expf(-(float)i * 0.57564627325f);  // ln(10000)/16
        float a  = (float)t * fr;
        sv[i] = __sinf(a);
        cv[i] = __cosf(a);
    }
    float* fsr = &fs_tab[(h * kN + t) * 16];
    float* fcr = &fc_tab[(h * kN + t) * 16];
#pragma unroll
    for (int d = 0; d < kOUTD; ++d) {
        float ss = 0.f, cs = 0.f, fs = 0.f, fc = 0.f;
#pragma unroll
        for (int D = 0; D < kIND; ++D) {
            float sc = sin_coef[(h * kIND + D) * kOUTD + d];
            float cc = cos_coef[(h * kIND + D) * kOUTD + d];
            ss += sc;  cs += cc;
            fs += sv[D] * sc;  fc += cv[D] * cc;
        }
        fsr[d] = fs / ss;
        fcr[d] = fc / cs;
    }
#pragma unroll
    for (int d = kOUTD; d < 16; ++d) { fsr[d] = 1.f; fcr[d] = 1.f; }
}

// =====================================================================
// K1: v = x @ v_w^T + v_b   (M=B*N, K=192, N=96) via WMMA
// =====================================================================
__global__ void vproj_kernel(const float* __restrict__ x,
                             const float* __restrict__ v_w,
                             const float* __restrict__ v_b,
                             float* __restrict__ v_buf) {
    int gw = blockIdx.x * 8 + (threadIdx.x >> 5);  // global wave == tile id
    int rt = gw / 6, ct = gw % 6;
    int m0 = rt * 16, c0 = ct * 16;
    v8f acc = {};
#pragma unroll
    for (int kk = 0; kk < 6; ++kk) {
        v16h a  = ldA(&x[(size_t)m0 * kDIM + kk * 32], kDIM);
        v16h bb = ldBt(&v_w[c0 * kDIM + kk * 32], kDIM);
        acc = wmma16(a, bb, acc);
    }
    int lane = threadIdx.x & 31, n = lane & 15, half = lane >> 4;
#pragma unroll
    for (int r = 0; r < 8; ++r) {
        int m = m0 + r + 8 * half;
        v_buf[(size_t)m * kC2 + c0 + n] = acc[r] + v_b[c0 + n];
    }
}

// =====================================================================
// K2 (k-side, fused): hedgehog(k) -> km partials + FoPE -> kv partials
// one block = 128 tokens of one (b,h); 4 waves do K-split WMMA then LDS-reduce
// =====================================================================
__global__ void kside_kernel(const float* __restrict__ k_in,
                             const float* __restrict__ v_buf,
                             const float* __restrict__ lnk_w,
                             const float* __restrict__ lnk_b,
                             const float* __restrict__ fs_tab,
                             const float* __restrict__ fc_tab,
                             float* __restrict__ km_buf,
                             float* __restrict__ kv_buf) {
    __shared__ float w_s[256], b_s[16], km_s[32];
    __shared__ float kfT[32][132];   // A: [feature f][token]
    __shared__ float vs[128][16];    // B: [token][d]
    __shared__ float part[4][512];   // per-wave kv partials
    int tid   = threadIdx.x;
    int chunk = blockIdx.x % 72;
    int bh    = blockIdx.x / 72;
    int b = bh / kNH, h = bh % kNH;
    int t0 = chunk * 128;

    w_s[tid]       = lnk_w[tid];
    w_s[tid + 128] = lnk_w[tid + 128];
    if (tid < 16) b_s[tid] = lnk_b[tid];
    if (tid < 32) km_s[tid] = 0.f;

    int t = t0 + tid;
    // v tile: async global->LDS (b128 x4 per token row of 16 floats)
#pragma unroll
    for (int d = 0; d < 16; d += 4)
        cp_b128(&v_buf[((size_t)b * kN + t) * kC2 + h * kHD + d], &vs[tid][d]);
    __syncthreads();

    float inp[16];
#pragma unroll
    for (int i = 0; i < 16; ++i)
        inp[i] = k_in[((size_t)b * kN + t) * kC2 + h * kHD + i];
    float fk[32], kf[32];
    hh_fope(inp, w_s, b_s, &fs_tab[(h * kN + t) * 16], &fc_tab[(h * kN + t) * 16],
            fk, kf);
#pragma unroll
    for (int j = 0; j < 32; ++j) atomicAdd(&km_s[j], fk[j]);
#pragma unroll
    for (int j = 0; j < 32; ++j) kfT[j][tid] = kf[j];
    cp_wait();
    __syncthreads();

    int wid = tid >> 5;  // wave handles K-chunk of 32 tokens
    v16h bb = ldB(&vs[wid * 32][0], 16);
    v16h a0 = ldA(&kfT[0][wid * 32], 132);
    v16h a1 = ldA(&kfT[16][wid * 32], 132);
    v8f acc0 = {}, acc1 = {};
    acc0 = wmma16(a0, bb, acc0);
    acc1 = wmma16(a1, bb, acc1);
    stD(&part[wid][0], 16, acc0);
    stD(&part[wid][256], 16, acc1);
    __syncthreads();

    if (tid < 32) atomicAdd(&km_buf[(b * kNH + h) * 32 + tid], km_s[tid]);
    for (int e = tid; e < 512; e += 128) {
        float s = part[0][e] + part[1][e] + part[2][e] + part[3][e];
        atomicAdd(&kv_buf[(b * kNH + h) * 512 + e], s);
    }
}

// =====================================================================
// K3 (q-side, fused): hedgehog(q) -> z from km -> FoPE -> xs = (qf@kv)*z
// writes xs straight into spatial buffer [B,N,C2]
// =====================================================================
__global__ void qside_kernel(const float* __restrict__ q_in,
                             const float* __restrict__ lnq_w,
                             const float* __restrict__ lnq_b,
                             const float* __restrict__ fs_tab,
                             const float* __restrict__ fc_tab,
                             const float* __restrict__ km_buf,
                             const float* __restrict__ kv_buf,
                             float* __restrict__ spatial) {
    __shared__ float w_s[256], b_s[16], km_s[32], kv_s[512];
    __shared__ float qfA[128][34];
    __shared__ float z_s[128];
    int tid   = threadIdx.x;
    int chunk = blockIdx.x % 72;
    int bh    = blockIdx.x / 72;
    int b = bh / kNH, h = bh % kNH;
    int t0 = chunk * 128;
    const float invN = 1.f / (float)kN;

    w_s[tid]       = lnq_w[tid];
    w_s[tid + 128] = lnq_w[tid + 128];
    if (tid < 16) b_s[tid] = lnq_b[tid];
    if (tid < 32) km_s[tid] = km_buf[(b * kNH + h) * 32 + tid] * invN;
    for (int e = tid; e < 512; e += 128)
        kv_s[e] = kv_buf[(b * kNH + h) * 512 + e] * invN;
    __syncthreads();

    int t = t0 + tid;
    float inp[16];
#pragma unroll
    for (int i = 0; i < 16; ++i)
        inp[i] = q_in[((size_t)b * kN + t) * kC2 + h * kHD + i];
    float fq[32], qf[32];
    hh_fope(inp, w_s, b_s, &fs_tab[(h * kN + t) * 16], &fc_tab[(h * kN + t) * 16],
            fq, qf);
    float z = 0.f;
#pragma unroll
    for (int j = 0; j < 32; ++j) z += fq[j] * km_s[j];
    z_s[tid] = 1.f / (z + 1e-6f);
#pragma unroll
    for (int j = 0; j < 32; ++j) qfA[tid][j] = qf[j];
    __syncthreads();

    int wid  = tid >> 5;
    int lane = tid & 31, n = lane & 15, half = lane >> 4;
    v16h bb = ldB(kv_s, 16);
#pragma unroll
    for (int si = 0; si < 2; ++si) {
        int s  = wid + si * 4;                 // 8 sub-tiles of 16 tokens
        v16h a = ldA(&qfA[s * 16][0], 34);
        v8f acc = {};
        acc = wmma16(a, bb, acc);
#pragma unroll
        for (int r = 0; r < 8; ++r) {
            int m = s * 16 + r + 8 * half;
            spatial[((size_t)b * kN + t0 + m) * kC2 + h * kHD + n] =
                acc[r] * z_s[m];
        }
    }
}

// =====================================================================
// K4: depthwise 3x3 LePE over v image, accumulate into spatial
// =====================================================================
__global__ void lepe_kernel(const float* __restrict__ v_buf,
                            const float* __restrict__ lepe_w,
                            const float* __restrict__ lepe_b,
                            float* __restrict__ spatial) {
    size_t idx = (size_t)blockIdx.x * 256 + threadIdx.x;  // over B*N*C2
    int c = (int)(idx % kC2);
    int n = (int)((idx / kC2) % kN);
    int b = (int)(idx / ((size_t)kC2 * kN));
    int y = n / kW, x = n % kW;
    float s = lepe_b[c];
#pragma unroll
    for (int ky = 0; ky < 3; ++ky) {
#pragma unroll
        for (int kx = 0; kx < 3; ++kx) {
            int yy = y + ky - 1, xx = x + kx - 1;
            if (yy >= 0 && yy < kW && xx >= 0 && xx < kW)
                s += v_buf[((size_t)b * kN + yy * kW + xx) * kC2 + c] *
                     lepe_w[c * 9 + ky * 3 + kx];
        }
    }
    spatial[idx] += s;
}

// =====================================================================
// K5: per-(b,c) L2 norms of q and k over tokens
// =====================================================================
__global__ void norm_kernel(const float* __restrict__ q_in,
                            const float* __restrict__ k_in,
                            float* __restrict__ qn, float* __restrict__ kn) {
    __shared__ float sq[256], sk[256];
    int blk = blockIdx.x;
    int b = blk / kC2, c = blk % kC2;
    int tid = threadIdx.x;
    float aq = 0.f, ak = 0.f;
    for (int n = tid; n < kN; n += 256) {
        float qv = q_in[((size_t)b * kN + n) * kC2 + c];
        float kv = k_in[((size_t)b * kN + n) * kC2 + c];
        aq += qv * qv;
        ak += kv * kv;
    }
    sq[tid] = aq; sk[tid] = ak;
    __syncthreads();
    for (int s = 128; s > 0; s >>= 1) {
        if (tid < s) { sq[tid] += sq[tid + s]; sk[tid] += sk[tid + s]; }
        __syncthreads();
    }
    if (tid == 0) {
        qn[blk] = rsqrtf(sq[0] + 1e-24f);
        kn[blk] = rsqrtf(sk[0] + 1e-24f);
    }
}

// =====================================================================
// K6: channel attention  S = (q^T k) scaled, softmax over d -> attn[b][96][96]
// one block per batch; 8 waves x up-to-5 tiles, K staged 32 tokens at a time
// (async global->LDS staging + next-chunk prefetch)
// =====================================================================
__global__ void chattn_kernel(const float* __restrict__ q_in,
                              const float* __restrict__ k_in,
                              const float* __restrict__ qn,
                              const float* __restrict__ kn,
                              const float* __restrict__ temperature,
                              float* __restrict__ attn_buf) {
    __shared__ float qT[96][33];     // [c][token-in-chunk]
    __shared__ float ks[32][100];    // [token-in-chunk][d]
    __shared__ float Sbuf[96][97];
    int b = blockIdx.x;
    int tid = threadIdx.x, wid = tid >> 5;
    v8f acc[5];
#pragma unroll
    for (int i = 0; i < 5; ++i) acc[i] = (v8f){};

    for (int chunk = 0; chunk < 288; ++chunk) {
        int n0 = chunk * 32;
        __syncthreads();
        // stage q chunk transposed: qT[c][kl] (strided gather -> b32 async)
        for (int i = tid; i < 3072; i += 256) {
            int c = i >> 5, kl = i & 31;
            cp_b32(&q_in[((size_t)b * kN + n0 + kl) * kC2 + c], &qT[c][kl]);
        }
        // stage k chunk row-major: ks[kl][d] (contiguous -> b128 async)
        for (int i = tid * 4; i < 3072; i += 1024) {
            int kl = i / 96, d = i % 96;
            cp_b128(&k_in[((size_t)b * kN + n0 + kl) * kC2 + d], &ks[kl][d]);
        }
        // prefetch next chunk of k into L2 while this chunk computes
        if (chunk + 1 < 288) {
            const float* nk = &k_in[((size_t)b * kN + n0 + 32) * kC2];
            __builtin_prefetch(nk + tid * 12, 0, 1);
        }
        cp_wait();
        __syncthreads();
#pragma unroll
        for (int i = 0; i < 5; ++i) {
            int tmix = wid + i * 8;
            if (tmix < 36) {
                int ct = tmix / 6, dt = tmix % 6;
                v16h a  = ldA(&qT[ct * 16][0], 33);
                v16h bb = ldB(&ks[0][dt * 16], 100);
                acc[i] = wmma16(a, bb, acc[i]);
            }
        }
    }
    int lane = tid & 31, n = lane & 15, half = lane >> 4;
#pragma unroll
    for (int i = 0; i < 5; ++i) {
        int tmix = wid + i * 8;
        if (tmix < 36) {
            int ct = tmix / 6, dt = tmix % 6;
#pragma unroll
            for (int r = 0; r < 8; ++r) {
                int c = ct * 16 + r + 8 * half;
                int d = dt * 16 + n;
                Sbuf[c][d] = acc[i][r] * qn[b * kC2 + c] * kn[b * kC2 + d] *
                             temperature[c];
            }
        }
    }
    __syncthreads();
    if (tid < 96) {
        int c = tid;
        float mx = -1e30f;
        for (int d = 0; d < 96; ++d) mx = fmaxf(mx, Sbuf[c][d]);
        float s = 0.f;
        for (int d = 0; d < 96; ++d) s += __expf(Sbuf[c][d] - mx);
        float inv = 1.f / s;
        for (int d = 0; d < 96; ++d)
            attn_buf[((size_t)b * kC2 + c) * kC2 + d] = __expf(Sbuf[c][d] - mx) * inv;
    }
}

// =====================================================================
// K7: M2[b][e][o] = sum_d attn[b][d][e] * proj_w[o][96+d]   (folds channel GEMM
// into the output projection)
// =====================================================================
__global__ void m2_kernel(const float* __restrict__ attn_buf,
                          const float* __restrict__ proj_w,
                          float* __restrict__ M2_buf) {
    __shared__ float aT[96][100];   // [e][d] = attn[b][d][e]
    int b = blockIdx.x;
    int tid = threadIdx.x, wid = tid >> 5;
    for (int i = tid; i < 9216; i += 256) {
        int d = i / 96, e = i % 96;
        cp_b32(&attn_buf[((size_t)b * kC2 + d) * kC2 + e], &aT[e][d]);
    }
    cp_wait();
    __syncthreads();
    int lane = tid & 31, n = lane & 15, half = lane >> 4;
    for (int t = wid; t < 72; t += 8) {
        int et = t / 12, ot = t % 12;
        v8f acc = {};
#pragma unroll
        for (int kk = 0; kk < 3; ++kk) {
            v16h a  = ldA(&aT[et * 16][kk * 32], 100);
            v16h bb = ldBt(&proj_w[(ot * 16) * kDIM + kC2 + kk * 32], kDIM);
            acc = wmma16(a, bb, acc);
        }
#pragma unroll
        for (int r = 0; r < 8; ++r) {
            int e = et * 16 + r + 8 * half;
            M2_buf[((size_t)b * kC2 + e) * kDIM + ot * 16 + n] = acc[r];
        }
    }
}

// =====================================================================
// K8: out = spatial @ proj_w[:, :96]^T + v @ M2[b] + proj_b
// =====================================================================
__global__ void finalproj_kernel(const float* __restrict__ spatial,
                                 const float* __restrict__ v_buf,
                                 const float* __restrict__ M2_buf,
                                 const float* __restrict__ proj_w,
                                 const float* __restrict__ proj_b,
                                 float* __restrict__ out) {
    int gw = blockIdx.x * 8 + (threadIdx.x >> 5);
    int rt = gw / 12, ct = gw % 12;
    int m0 = rt * 16, o0 = ct * 16;
    int b = m0 / kN;
    v8f acc = {};
#pragma unroll
    for (int kk = 0; kk < 3; ++kk) {   // spatial part: K = 0..95
        v16h a  = ldA(&spatial[(size_t)m0 * kC2 + kk * 32], kC2);
        v16h bb = ldBt(&proj_w[o0 * kDIM + kk * 32], kDIM);
        acc = wmma16(a, bb, acc);
    }
#pragma unroll
    for (int kk = 0; kk < 3; ++kk) {   // channel part folded through M2
        v16h a  = ldA(&v_buf[(size_t)m0 * kC2 + kk * 32], kC2);
        v16h bb = ldB(&M2_buf[((size_t)b * kC2 + kk * 32) * kDIM + o0], kDIM);
        acc = wmma16(a, bb, acc);
    }
    int lane = threadIdx.x & 31, n = lane & 15, half = lane >> 4;
#pragma unroll
    for (int r = 0; r < 8; ++r) {
        int m = m0 + r + 8 * half;
        out[(size_t)m * kDIM + o0 + n] = acc[r] + proj_b[o0 + n];
    }
}

// =====================================================================
// launch
// =====================================================================
extern "C" void kernel_launch(void* const* d_in, const int* in_sizes, int n_in,
                              void* d_out, int out_size, void* d_ws, size_t ws_size,
                              hipStream_t stream) {
    (void)in_sizes; (void)n_in; (void)out_size; (void)ws_size;
    const float* x        = (const float*)d_in[0];
    const float* q        = (const float*)d_in[1];
    const float* k        = (const float*)d_in[2];
    const float* v_w      = (const float*)d_in[3];
    const float* v_b      = (const float*)d_in[4];
    const float* lnq_w    = (const float*)d_in[5];
    const float* lnq_b    = (const float*)d_in[6];
    const float* lnk_w    = (const float*)d_in[7];
    const float* lnk_b    = (const float*)d_in[8];
    const float* sin_coef = (const float*)d_in[9];
    const float* cos_coef = (const float*)d_in[10];
    const float* lepe_w   = (const float*)d_in[11];
    const float* lepe_b   = (const float*)d_in[12];
    const float* proj_w   = (const float*)d_in[13];
    const float* proj_b   = (const float*)d_in[14];
    const float* temp     = (const float*)d_in[15];
    float*       out      = (float*)d_out;

    // workspace carve-up (floats)
    float* ws = (float*)d_ws;
    size_t o = 0;
    float* fs_tab  = ws + o; o += (size_t)kNH * kN * 16;
    float* fc_tab  = ws + o; o += (size_t)kNH * kN * 16;
    float* v_buf   = ws + o; o += (size_t)kB * kN * kC2;
    float* spatial = ws + o; o += (size_t)kB * kN * kC2;
    float* km_buf  = ws + o; o += (size_t)kB * kNH * 32;
    float* kv_buf  = ws + o; o += (size_t)kB * kNH * 512;
    float* qn      = ws + o; o += (size_t)kB * kC2;
    float* kn      = ws + o; o += (size_t)kB * kC2;
    float* attn    = ws + o; o += (size_t)kB * kC2 * kC2;
    float* M2      = ws + o; o += (size_t)kB * kC2 * kDIM;

    // km/kv accumulated with atomics -> zero them (graph-capture-safe memset)
    (void)hipMemsetAsync(km_buf, 0,
                         (size_t)(kB * kNH * 32 + kB * kNH * 512) * sizeof(float),
                         stream);

    // K0: FoPE tables
    fope_tab_kernel<<<(kNH * kN + 255) / 256, 256, 0, stream>>>(sin_coef, cos_coef,
                                                                fs_tab, fc_tab);
    // K1: v projection  (55296 tiles / 8 waves)
    vproj_kernel<<<6912, 256, 0, stream>>>(x, v_w, v_b, v_buf);
    // K2: k-side fused (km + kv)
    kside_kernel<<<kB * kNH * 72, 128, 0, stream>>>(k, v_buf, lnk_w, lnk_b, fs_tab,
                                                    fc_tab, km_buf, kv_buf);
    // K3: q-side fused -> xs (into spatial)
    qside_kernel<<<kB * kNH * 72, 128, 0, stream>>>(q, lnq_w, lnq_b, fs_tab, fc_tab,
                                                    km_buf, kv_buf, spatial);
    // K4: LePE depthwise conv accumulate
    lepe_kernel<<<(int)(((size_t)kB * kN * kC2) / 256), 256, 0, stream>>>(
        v_buf, lepe_w, lepe_b, spatial);
    // K5: channel L2 norms
    norm_kernel<<<kB * kC2, 256, 0, stream>>>(q, k, qn, kn);
    // K6: channel attention softmax
    chattn_kernel<<<kB, 256, 0, stream>>>(q, k, qn, kn, temp, attn);
    // K7: fold attn into projection weights
    m2_kernel<<<kB, 256, 0, stream>>>(attn, proj_w, M2);
    // K8: fused output projection (spatial + channel branches)
    finalproj_kernel<<<((kB * kN / 16) * 12) / 8, 256, 0, stream>>>(
        spatial, v_buf, M2, proj_w, proj_b, out);
}
